// TransducerJointNetworkV1_77859167141918
// MI455X (gfx1250) — matrix operand order repaired and checked by reading, not compiled, not run on hardware
//
#include <hip/hip_runtime.h>

typedef __attribute__((ext_vector_type(16))) __bf16        v16bf;
typedef __attribute__((ext_vector_type(8)))  float         v8f;
typedef __attribute__((ext_vector_type(8)))  unsigned int  v8u;

union FragCast { v8u u; v16bf b; };

#define F_ENC 512
#define H     1024
#define V     1024
#define Ssz   64
#define KC    32
#define HPAD  40   // bf16 elems per padded LDS row (80 B) -> conflict-free b128 reads

__device__ __forceinline__ unsigned int pack_bf16x2(float lo, float hi) {
  unsigned int a = __float_as_uint(lo);
  unsigned int b = __float_as_uint(hi);
  a += 0x7fffu + ((a >> 16) & 1u);          // round-to-nearest-even
  b += 0x7fffu + ((b >> 16) & 1u);
  return (a >> 16) | (b & 0xffff0000u);
}

// ---------------- Projection: P = src@Ws + b1, Q = tgt@Wt (fp32) ----------------
__global__ __launch_bounds__(256) void proj_kernel(
    const float* __restrict__ src, const float* __restrict__ tgt,
    const float* __restrict__ W1, const float* __restrict__ b1,
    float* __restrict__ P, float* __restrict__ Q)
{
  __shared__ __align__(16) float erow[8 * F_ENC];  // 8 encoding rows, 16 KB
  const int tid = threadIdx.x;
  const int rb  = blockIdx.x;                      // 0..191
  const bool is_src = rb < 128;
  const int row0 = is_src ? rb * 8 : (rb - 128) * 8;
  const float* encBase = is_src ? (src + (size_t)row0 * F_ENC)
                                : (tgt + (size_t)row0 * F_ENC);
  const float* Wb = is_src ? W1 : (W1 + (size_t)F_ENC * H);
  const int col = blockIdx.y * 256 + tid;

  for (int i = tid; i < 8 * F_ENC; i += 256) erow[i] = encBase[i];
  __syncthreads();

  float acc[8];
  const float binit = is_src ? b1[col] : 0.0f;
  #pragma unroll
  for (int r = 0; r < 8; ++r) acc[r] = binit;

  #pragma unroll 4
  for (int k = 0; k < F_ENC; ++k) {
    float w = Wb[(size_t)k * H + col];
    #pragma unroll
    for (int r = 0; r < 8; ++r) acc[r] += erow[r * F_ENC + k] * w;
  }

  float* dst = is_src ? (P + (size_t)row0 * H + col) : (Q + (size_t)row0 * H + col);
  #pragma unroll
  for (int r = 0; r < 8; ++r) dst[(size_t)r * H] = acc[r];
}

// ---------------- W2 (H x V fp32) -> W2T (V x H bf16), LDS-tiled transpose ----------------
__global__ __launch_bounds__(256) void w2t_kernel(const float* __restrict__ W2,
                                                  unsigned short* __restrict__ W2T)
{
  __shared__ unsigned short tile[64][68];
  const int tid = threadIdx.x;
  const int h0 = blockIdx.x * 64;
  const int v0 = blockIdx.y * 64;
  for (int i = tid; i < 64 * 64; i += 256) {
    int r = i >> 6, c = i & 63;                              // r=h, c=v (coalesced read)
    unsigned int u = __float_as_uint(W2[(size_t)(h0 + r) * V + (v0 + c)]);
    u += 0x7fffu + ((u >> 16) & 1u);
    tile[c][r] = (unsigned short)(u >> 16);
  }
  __syncthreads();
  for (int i = tid; i < 64 * 64; i += 256) {
    int r = i >> 6, c = i & 63;                              // r=v, c=h (coalesced write)
    W2T[(size_t)(v0 + r) * H + (h0 + c)] = tile[r][c];
  }
}

// ---------------- Fused joint: out[bt, s, v] = relu(P[bt]+Q[b,s]) @ W2 + b2 ----------------
__global__ __launch_bounds__(256) void joint_kernel(
    const float* __restrict__ P, const float* __restrict__ Q,
    const unsigned short* __restrict__ W2T, const float* __restrict__ b2,
    float* __restrict__ out)
{
  __shared__ __align__(16) float Pb[H];                          // 4 KB
  __shared__ __align__(16) unsigned short hbuf[2][Ssz * HPAD];   // 2 x 5 KB ping-pong

  const int tid  = threadIdx.x;
  const int bt   = blockIdx.x;        // 0..1023 (b*T + t)
  const int b    = bt >> 7;           // T = 128
  const int vblk = blockIdx.y;        // 0..3 (256 cols each)
  const int wave = tid >> 5;
  const int lane = tid & 31;
  const int lanelo = lane & 15;
  const int lanehi = lane >> 4;

  { // preload src projection row (already includes b1)
    const float* prow = P + (size_t)bt * H;
    for (int i = tid; i < H; i += 256) Pb[i] = prow[i];
  }

  // producer mapping: each thread owns (s, 8 consecutive k) of the 64x32 h tile
  const int s_p = tid >> 2;
  const int kq  = (tid & 3) * 8;
  const float* qrow = Q + ((size_t)(b * Ssz + s_p)) * H + kq;

  // consumer mapping: 8 waves -> 4 M-strips x 2 N-halves; 8 16x16 tiles per wave
  const int mstrip = wave & 3;
  const int nhalf  = wave >> 2;
  const int nbase  = vblk * 256 + nhalf * 128;
  const int aoff   = (mstrip * 16 + lanelo) * HPAD + lanehi * 8;  // ISA A layout

  const unsigned short* w2p[8];
  float bias[8];
  #pragma unroll
  for (int t = 0; t < 8; ++t) {
    int vcol = nbase + t * 16 + lanelo;
    w2p[t]  = W2T + (size_t)vcol * H + lanehi * 8;                // ISA B layout
    bias[t] = b2[vcol];
  }

  v8f acc[8];
  #pragma unroll
  for (int t = 0; t < 8; ++t) acc[t] = (v8f)(0.0f);

  __syncthreads();  // Pb ready

  int ib = 0;
  for (int kg = 0; kg < H; kg += KC, ib ^= 1) {
    // ---- produce 64x32 bf16 tile: relu(P + Q) ----
    float4 q0 = *(const float4*)(qrow + kg);
    float4 q1 = *(const float4*)(qrow + kg + 4);
    const float4* pb = (const float4*)(&Pb[kg + kq]);
    float4 p0 = pb[0], p1 = pb[1];
    uint4 hp;
    hp.x = pack_bf16x2(fmaxf(p0.x + q0.x, 0.f), fmaxf(p0.y + q0.y, 0.f));
    hp.y = pack_bf16x2(fmaxf(p0.z + q0.z, 0.f), fmaxf(p0.w + q0.w, 0.f));
    hp.z = pack_bf16x2(fmaxf(p1.x + q1.x, 0.f), fmaxf(p1.y + q1.y, 0.f));
    hp.w = pack_bf16x2(fmaxf(p1.z + q1.z, 0.f), fmaxf(p1.w + q1.w, 0.f));
    *(uint4*)(&hbuf[ib][s_p * HPAD + kq]) = hp;
    __syncthreads();

    // ---- consume: A fragment from LDS (lane<16: K 0-7,16-23; lane>=16: K 8-15,24-31)
    const uint4* ap = (const uint4*)(&hbuf[ib][aoff]);
    uint4 alo = ap[0];
    uint4 ahi = ap[2];   // +16 bf16 elems
    FragCast a;
    a.u[0]=alo.x; a.u[1]=alo.y; a.u[2]=alo.z; a.u[3]=alo.w;
    a.u[4]=ahi.x; a.u[5]=ahi.y; a.u[6]=ahi.z; a.u[7]=ahi.w;

    #pragma unroll
    for (int t = 0; t < 8; ++t) {
      const uint4* bp = (const uint4*)(w2p[t] + kg);
      uint4 blo = bp[0], bhi = bp[2];
      FragCast bb;
      bb.u[0]=blo.x; bb.u[1]=blo.y; bb.u[2]=blo.z; bb.u[3]=blo.w;
      bb.u[4]=bhi.x; bb.u[5]=bhi.y; bb.u[6]=bhi.z; bb.u[7]=bhi.w;
      acc[t] = __builtin_amdgcn_wmma_f32_16x16x32_bf16(
          false, a.b, false, bb.b, (short)0, acc[t], false, false);
    }
  }

  // ---- epilogue: C layout is VGPR r -> row r (lanes 0-15) / row r+8 (lanes 16-31)
  const int srow = mstrip * 16 + lanehi * 8;
  #pragma unroll
  for (int t = 0; t < 8; ++t) {
    const int v = nbase + t * 16 + lanelo;
    const float bs = bias[t];
    #pragma unroll
    for (int r = 0; r < 8; ++r) {
      out[((size_t)(bt * Ssz + srow + r)) * V + v] = acc[t][r] + bs;
    }
  }
}

extern "C" void kernel_launch(void* const* d_in, const int* in_sizes, int n_in,
                              void* d_out, int out_size, void* d_ws, size_t ws_size,
                              hipStream_t stream) {
  (void)in_sizes; (void)n_in; (void)out_size; (void)ws_size;
  const float* src = (const float*)d_in[0];
  const float* tgt = (const float*)d_in[1];
  const float* W1  = (const float*)d_in[2];
  const float* b1  = (const float*)d_in[3];
  const float* W2  = (const float*)d_in[4];
  const float* b2  = (const float*)d_in[5];
  float* out = (float*)d_out;

  char* ws = (char*)d_ws;
  float* P = (float*)ws;                                             // 1024x1024 f32 (4 MB)
  float* Q = (float*)(ws + (size_t)1024 * 1024 * 4);                 // 512x1024 f32 (2 MB)
  unsigned short* W2T =
      (unsigned short*)(ws + (size_t)1024 * 1024 * 4 + (size_t)512 * 1024 * 4); // 2 MB bf16

  proj_kernel <<<dim3(192, 4),  256, 0, stream>>>(src, tgt, W1, b1, P, Q);
  w2t_kernel  <<<dim3(16, 16),  256, 0, stream>>>(W2, W2T);
  joint_kernel<<<dim3(1024, 4), 256, 0, stream>>>(P, Q, W2T, b2, out);
}
